// MultiheadSelfAttention_61942018343119
// MI455X (gfx1250) — compile-verified, matrix operand order checked by reading, not run on hardware
//
#include <hip/hip_runtime.h>

typedef __bf16 bf16;
typedef __attribute__((ext_vector_type(16))) __bf16 v16bf;
typedef __attribute__((ext_vector_type(4)))  __bf16 v4bf;
typedef __attribute__((ext_vector_type(8)))  float  v8f;
typedef __attribute__((ext_vector_type(4)))  int    v4i;
typedef __attribute__((ext_vector_type(8)))  int    v8i;
typedef __attribute__((ext_vector_type(4)))  unsigned int v4u;

#define BATCH  2
#define SEQ    2048
#define DMODEL 1024
#define HEADS  16
#define DK     64
#define MTOT   (BATCH*SEQ)

// ---- feature probes (graceful fallback keeps compile green) ----------------
#if __has_builtin(__builtin_amdgcn_global_load_async_to_lds_b128) && \
    __has_builtin(__builtin_amdgcn_s_wait_asynccnt)
#define HAVE_ASYNC 1
#else
#define HAVE_ASYNC 0
#endif

#if __has_builtin(__builtin_amdgcn_tensor_load_to_lds) && \
    __has_builtin(__builtin_amdgcn_s_wait_tensorcnt)
#define HAVE_TDM 1
#else
#define HAVE_TDM 0
#endif

#if __has_builtin(__builtin_amdgcn_update_dpp) && __has_builtin(__builtin_amdgcn_mov_dpp8)
#define HAVE_DPP 1
#else
#define HAVE_DPP 0
#endif

// ---------------------------------------------------------------- WMMA helper
__device__ __forceinline__ v8f wmma_bf16(v16bf a, v16bf b, v8f c) {
  return __builtin_amdgcn_wmma_f32_16x16x32_bf16(
      /*neg_a=*/false, a, /*neg_b=*/false, b,
      /*c_mod=*/(short)0, c, /*reuse_a=*/false, /*reuse_b=*/false);
}

union FragU { v16bf v; uint4 q[2]; };

__device__ __forceinline__ v16bf load_frag(const bf16* p) {
  FragU u;
  u.q[0] = *(const uint4*)(p);
  u.q[1] = *(const uint4*)(p + 16);
  return u.v;
}

__device__ __forceinline__ v8f vzero8() {
  v8f z;
#pragma unroll
  for (int r = 0; r < 8; ++r) z[r] = 0.0f;
  return z;
}

// ---- async global->LDS copy (16 bytes / lane) ------------------------------
#if HAVE_ASYNC
__device__ __forceinline__ void async_cp16(bf16* lds, const bf16* g) {
  __builtin_amdgcn_global_load_async_to_lds_b128(
      (__attribute__((address_space(1))) v4i*)g,
      (__attribute__((address_space(3))) v4i*)lds,
      /*imm offset*/0, /*cpol*/0);
}
#endif

__device__ __forceinline__ void copy16_sync(bf16* lds, const bf16* g) {
  uint4 v = *(const uint4*)g;
  *(uint4*)lds = v;
}

// ---- TDM: 2D tile load global -> LDS (D# per ISA 8.3-8.6) ------------------
#if HAVE_TDM
__device__ __forceinline__ unsigned lds_off(const void* p) {
  return (unsigned)(unsigned long long)
      (__attribute__((address_space(3))) const void*)p;
}

// rows x row_dwords tile of 4-byte units; row stride in dwords; LDS rows get
// pad_amount(code) dwords of padding after every pad_interval(code) dwords.
__device__ __forceinline__ void tdm_load_2d(const bf16* gsrc, unsigned ldsaddr,
                                            unsigned rows, unsigned row_dwords,
                                            unsigned stride_dwords,
                                            int pad_interval, int pad_amount) {
  unsigned long long ga = (unsigned long long)gsrc;
  v4u g0;
  g0[0] = 1u;                                   // count=1 (valid), user mode
  g0[1] = ldsaddr;                              // LDS byte address
  g0[2] = (unsigned)(ga & 0xffffffffu);         // global addr [31:0]
  g0[3] = (unsigned)((ga >> 32) & 0x1ffffffu)   // global addr [56:32]
          | (2u << 30);                         // type = 2 (image)
  v8i g1;
  // data_size=2 (4B), pad_enable=1, pad_interval/pad_amount codes
  g1[0] = (int)((2u << 16) | (1u << 20) |
                ((unsigned)pad_interval << 22) | ((unsigned)pad_amount << 25));
  const unsigned td0 = 0x40000000u, td1 = 0x40000000u;   // huge (no clipping)
  g1[1] = (int)((td0 & 0xffffu) << 16);                      // tensor_dim0 lo
  g1[2] = (int)(((td0 >> 16) & 0xffffu) | ((td1 & 0xffffu) << 16));
  g1[3] = (int)(((td1 >> 16) & 0xffffu) | (row_dwords << 16)); // tile_dim0
  g1[4] = (int)(rows & 0xffffu);                             // tile_dim1, dim2=0
  g1[5] = (int)stride_dwords;                                // dim0_stride[31:0]
  g1[6] = 0;                                                 // hi bits, dim1_stride
  g1[7] = 0;
  v4i z4;
  z4[0] = 0; z4[1] = 0; z4[2] = 0; z4[3] = 0;
  v8i z8;
#pragma unroll
  for (int i = 0; i < 8; ++i) z8[i] = 0;
  // 6-arg flavor (clang-23 / therock headers): extra int32x8 before cpol
  __builtin_amdgcn_tensor_load_to_lds(g0, g1, z4, z4, z8, /*cpol*/0);
}
#endif

// ---- xor-butterfly within a 16-lane half via DPP (fallback: ds_bpermute) ---
#if HAVE_DPP
constexpr unsigned dpp8_xor_sel(int m) {
  unsigned r = 0;
  for (int i = 0; i < 8; ++i) r |= (unsigned)((i ^ m) & 7) << (3 * i);
  return r;
}
#endif

template <int M>
__device__ __forceinline__ float lane_xor_f(float x) {
#if HAVE_DPP
  int xi = __builtin_bit_cast(int, x);
  int r;
  if constexpr (M == 1) {        // quad_perm [1,0,3,2]
    r = __builtin_amdgcn_update_dpp(0, xi, 0xB1, 0xF, 0xF, true);
  } else if constexpr (M == 2) { // quad_perm [2,3,0,1]
    r = __builtin_amdgcn_update_dpp(0, xi, 0x4E, 0xF, 0xF, true);
  } else if constexpr (M == 4) { // DPP8 permutation i^4
    constexpr unsigned S = dpp8_xor_sel(4);
    r = __builtin_amdgcn_mov_dpp8(xi, S);
  } else {                       // xor8 == row_ror:8 within a 16-lane row
    r = __builtin_amdgcn_update_dpp(0, xi, 0x128, 0xF, 0xF, true);
  }
  return __builtin_bit_cast(float, r);
#else
  return __shfl_xor(x, M, 32);
#endif
}

__device__ __forceinline__ float half_reduce_max(float v) {
  v = fmaxf(v, lane_xor_f<1>(v));
  v = fmaxf(v, lane_xor_f<2>(v));
  v = fmaxf(v, lane_xor_f<4>(v));
  v = fmaxf(v, lane_xor_f<8>(v));
  return v;
}
__device__ __forceinline__ float half_reduce_sum(float v) {
  v += lane_xor_f<1>(v);
  v += lane_xor_f<2>(v);
  v += lane_xor_f<4>(v);
  v += lane_xor_f<8>(v);
  return v;
}

// ---------------------------------------------------------------- f32 -> bf16
__global__ void cast_kernel(const float* __restrict__ in, bf16* __restrict__ out, int n4) {
  int i = blockIdx.x * blockDim.x + threadIdx.x;
  if (i < n4) {
    float4 f = ((const float4*)in)[i];
    v4bf o;
    o.x = (bf16)f.x; o.y = (bf16)f.y; o.z = (bf16)f.z; o.w = (bf16)f.w;
    ((v4bf*)out)[i] = o;
  }
}

// ---------------------------------------------------------------- GEMM (WMMA)
// C[m,n] = sum_k A[m,k] * W[n,k]
// MODE 0: bf16 -> [B,H,S,dk] | MODE 1: bf16 -> [B,H,dk,S] | MODE 2: f32 -> [M,D]
template <int MODE>
__global__ __launch_bounds__(256)
void gemm_wmma(const bf16* __restrict__ A, const bf16* __restrict__ W,
               bf16* __restrict__ outb, float* __restrict__ outf) {
  __shared__ bf16 As[2][128][40];   // double-buffered 128x32 tiles, 80B rows
  __shared__ bf16 Bs[2][128][40];

  const int tid  = threadIdx.x;
  const int lane = tid & 31;
  const int wid  = tid >> 5;
  const int wm   = wid >> 1;
  const int wn   = wid & 1;
  const int m0   = blockIdx.y * 128;
  const int n0   = blockIdx.x * 128;
  const int lrow = lane & 15;
  const int kb   = (lane < 16) ? 0 : 8;
  const int srow = tid >> 1;
  const int scol = (tid & 1) * 16;

  const bf16* gA = A + (size_t)(m0 + srow) * DMODEL + scol;
  const bf16* gB = W + (size_t)(n0 + srow) * DMODEL + scol;
  (void)gA; (void)gB;

#if HAVE_TDM
  const unsigned ldsA0 = lds_off(&As[0][0][0]);
  const unsigned ldsB0 = lds_off(&Bs[0][0][0]);
#endif

  auto stage = [&](int buf, int k0) {
#if HAVE_TDM
    if (wid == 0) {
      // 128 rows x 16 dwords, row stride 512 dwords; LDS pad 4dw after 16dw
      tdm_load_2d(A + (size_t)m0 * DMODEL + k0, ldsA0 + (unsigned)buf * 10240u,
                  128, 16, 512, 3, 3);
      tdm_load_2d(W + (size_t)n0 * DMODEL + k0, ldsB0 + (unsigned)buf * 10240u,
                  128, 16, 512, 3, 3);
    }
#elif HAVE_ASYNC
    bf16* la = &As[buf][srow][scol];
    bf16* lb = &Bs[buf][srow][scol];
    async_cp16(la,     gA + k0);
    async_cp16(la + 8, gA + k0 + 8);
    async_cp16(lb,     gB + k0);
    async_cp16(lb + 8, gB + k0 + 8);
#else
    bf16* la = &As[buf][srow][scol];
    bf16* lb = &Bs[buf][srow][scol];
    copy16_sync(la,     gA + k0);
    copy16_sync(la + 8, gA + k0 + 8);
    copy16_sync(lb,     gB + k0);
    copy16_sync(lb + 8, gB + k0 + 8);
#endif
  };

  v8f acc[2][4];
#pragma unroll
  for (int i = 0; i < 2; ++i)
#pragma unroll
    for (int j = 0; j < 4; ++j) acc[i][j] = vzero8();

  constexpr int NKT = DMODEL / 32;
  stage(0, 0);
  for (int kt = 0; kt < NKT; ++kt) {
    const int cur = kt & 1;
    if (kt + 1 < NKT) {
      stage(1 - cur, (kt + 1) * 32);
#if HAVE_TDM
      if (wid == 0) __builtin_amdgcn_s_wait_tensorcnt(2);
#elif HAVE_ASYNC
      __builtin_amdgcn_s_wait_asynccnt(4);
#endif
    } else {
#if HAVE_TDM
      if (wid == 0) __builtin_amdgcn_s_wait_tensorcnt(0);
#elif HAVE_ASYNC
      __builtin_amdgcn_s_wait_asynccnt(0);
#endif
    }
    __syncthreads();

    v16bf af[2], bfm[4];
#pragma unroll
    for (int ms = 0; ms < 2; ++ms)
      af[ms] = load_frag(&As[cur][wm * 32 + ms * 16 + lrow][kb]);
#pragma unroll
    for (int ns = 0; ns < 4; ++ns)
      bfm[ns] = load_frag(&Bs[cur][wn * 64 + ns * 16 + lrow][kb]);
#pragma unroll
    for (int ms = 0; ms < 2; ++ms)
#pragma unroll
      for (int ns = 0; ns < 4; ++ns)
        acc[ms][ns] = wmma_bf16(af[ms], bfm[ns], acc[ms][ns]);
    __syncthreads();
  }

  const int rowadd = (lane >= 16) ? 8 : 0;
#pragma unroll
  for (int ms = 0; ms < 2; ++ms) {
#pragma unroll
    for (int ns = 0; ns < 4; ++ns) {
#pragma unroll
      for (int r = 0; r < 8; ++r) {
        int m = m0 + wm * 32 + ms * 16 + r + rowadd;
        int n = n0 + wn * 64 + ns * 16 + (lane & 15);
        float v = acc[ms][ns][r];
        if (MODE == 2) {
          outf[(size_t)m * DMODEL + n] = v;
        } else {
          int b = m / SEQ, s = m % SEQ;
          int h = n / DK,  j = n % DK;
          size_t idx;
          if (MODE == 0) idx = ((size_t)(b * HEADS + h) * SEQ + s) * DK + j;
          else           idx = ((size_t)(b * HEADS + h) * DK + j) * SEQ + s;
          outb[idx] = (bf16)v;
        }
      }
    }
  }
}

// ---------------------------------------------------------------- RoPE in-place
__global__ void rope_kernel(bf16* __restrict__ T, const int* __restrict__ pos) {
  int gid = blockIdx.x * blockDim.x + threadIdx.x;
  int i  = gid & 31;
  int s  = (gid >> 5) & (SEQ - 1);
  int bh = gid >> 16;
  if (bh >= BATCH * HEADS) return;
  size_t base = ((size_t)bh * SEQ + s) * DK + 2 * i;
  float xr = (float)T[base];
  float xi = (float)T[base + 1];
  float fr = __expf(-(float)(2 * i) * (9.210340371976184f / 64.0f));
  float ang = (float)pos[s] * fr;
  float sn, cs;
  __sincosf(ang, &sn, &cs);
  T[base]     = (bf16)(xr * cs - xi * sn);
  T[base + 1] = (bf16)(xr * sn + xi * cs);
}

// ---------------------------------------------------------------- Flash attention
__global__ __launch_bounds__(128)
void attn_kernel(const bf16* __restrict__ Q, const bf16* __restrict__ K,
                 const bf16* __restrict__ Vt, bf16* __restrict__ Ab) {
  __shared__ bf16 Ks[2][32][72];     // double-buffered 32 keys x 64 dk
  __shared__ bf16 Vs[2][64][40];     // double-buffered 64 dk x 32 keys
  __shared__ bf16 Ps[4][16][40];     // per-wave P transpose buffer

  const int tid    = threadIdx.x;
  const int lane   = tid & 31;
  const int wid    = tid >> 5;
  const int qblk   = blockIdx.x;
  const int bh     = blockIdx.y;
  const int q0     = qblk * 64 + wid * 16;
  const int lrow   = lane & 15;
  const int kb     = (lane < 16) ? 0 : 8;
  const int rowadd = (lane >= 16) ? 8 : 0;

  const bf16* Qp = Q  + (size_t)bh * SEQ * DK;
  const bf16* Kp = K  + (size_t)bh * SEQ * DK;
  const bf16* Vp = Vt + (size_t)bh * DK * SEQ;

  // per-thread staging coordinates (fallback paths)
  const int krow = tid >> 2, kch = tid & 3;
  const int vrow = tid >> 1, vch = tid & 1;
  (void)krow; (void)kch; (void)vrow; (void)vch;

#if HAVE_TDM
  const unsigned ldsK0 = lds_off(&Ks[0][0][0]);
  const unsigned ldsV0 = lds_off(&Vs[0][0][0]);
#endif

  auto stage_kv = [&](int buf, int key0) {
#if HAVE_TDM
    if (wid == 0) {
      // K: 32 rows x 32 dwords (stride 32 dwords); pad 4dw after 32dw
      tdm_load_2d(Kp + (size_t)key0 * DK, ldsK0 + (unsigned)buf * 4608u,
                  32, 32, 32, 4, 3);
      // V: 64 rows x 16 dwords (stride 1024 dwords); pad 4dw after 16dw
      tdm_load_2d(Vp + key0, ldsV0 + (unsigned)buf * 5120u,
                  64, 16, 1024, 3, 3);
    }
#elif HAVE_ASYNC
    const bf16* gk = Kp + (size_t)(key0 + krow) * DK + kch * 16;
    const bf16* gv = Vp + (size_t)vrow * SEQ + key0 + vch * 16;
    bf16* lk = &Ks[buf][krow][kch * 16];
    bf16* lv = &Vs[buf][vrow][vch * 16];
    async_cp16(lk,     gk);
    async_cp16(lk + 8, gk + 8);
    async_cp16(lv,     gv);
    async_cp16(lv + 8, gv + 8);
#else
    const bf16* gk = Kp + (size_t)(key0 + krow) * DK + kch * 16;
    const bf16* gv = Vp + (size_t)vrow * SEQ + key0 + vch * 16;
    bf16* lk = &Ks[buf][krow][kch * 16];
    bf16* lv = &Vs[buf][vrow][vch * 16];
    copy16_sync(lk,     gk);
    copy16_sync(lk + 8, gk + 8);
    copy16_sync(lv,     gv);
    copy16_sync(lv + 8, gv + 8);
#endif
  };

  v16bf qf[2];
#pragma unroll
  for (int c = 0; c < 2; ++c)
    qf[c] = load_frag(Qp + (size_t)(q0 + lrow) * DK + c * 32 + kb);

  float mst[8], lst[8];
#pragma unroll
  for (int r = 0; r < 8; ++r) { mst[r] = -1e30f; lst[r] = 0.0f; }
  v8f o[4];
#pragma unroll
  for (int j = 0; j < 4; ++j) o[j] = vzero8();

  const int nkb = qblk * 2 + 2;
  stage_kv(0, 0);
  for (int kblk = 0; kblk < nkb; ++kblk) {
    const int cur  = kblk & 1;
    const int key0 = kblk * 32;
    if (kblk + 1 < nkb) {
      stage_kv(1 - cur, (kblk + 1) * 32);
#if HAVE_TDM
      if (wid == 0) __builtin_amdgcn_s_wait_tensorcnt(2);
#elif HAVE_ASYNC
      __builtin_amdgcn_s_wait_asynccnt(4);
#endif
    } else {
#if HAVE_TDM
      if (wid == 0) __builtin_amdgcn_s_wait_tensorcnt(0);
#elif HAVE_ASYNC
      __builtin_amdgcn_s_wait_asynccnt(0);
#endif
    }
    __syncthreads();

    if (key0 <= q0 + 15) {
      v16bf kf[2][2];
#pragma unroll
      for (int ns = 0; ns < 2; ++ns)
#pragma unroll
        for (int c = 0; c < 2; ++c)
          kf[c][ns] = load_frag(&Ks[cur][ns * 16 + lrow][c * 32 + kb]);

      v8f sc[2];
#pragma unroll
      for (int ns = 0; ns < 2; ++ns)
        sc[ns] = wmma_bf16(qf[1], kf[1][ns], wmma_bf16(qf[0], kf[0][ns], vzero8()));

      float p[2][8], rmax[8];
#pragma unroll
      for (int r = 0; r < 8; ++r) {
        int m = q0 + r + rowadd;
#pragma unroll
        for (int ns = 0; ns < 2; ++ns) {
          int key = key0 + ns * 16 + (lane & 15);
          float v = sc[ns][r] * 0.125f;
          p[ns][r] = (key <= m) ? v : -1e30f;
        }
        rmax[r] = half_reduce_max(fmaxf(p[0][r], p[1][r]));
      }
#pragma unroll
      for (int r = 0; r < 8; ++r) {
        float mnew = fmaxf(mst[r], rmax[r]);
        float corr = __expf(mst[r] - mnew);
        mst[r] = mnew;
        float rs = 0.0f;
#pragma unroll
        for (int ns = 0; ns < 2; ++ns) {
          float e = __expf(p[ns][r] - mnew);
          p[ns][r] = e;
          rs += e;
        }
        rs = half_reduce_sum(rs);
        lst[r] = lst[r] * corr + rs;
#pragma unroll
        for (int j = 0; j < 4; ++j) o[j][r] *= corr;
      }

      // C-layout -> A-fragment transpose through wave-private LDS (DS in-order)
#pragma unroll
      for (int r = 0; r < 8; ++r)
#pragma unroll
        for (int ns = 0; ns < 2; ++ns)
          Ps[wid][r + rowadd][ns * 16 + (lane & 15)] = (bf16)p[ns][r];

      v16bf pf = load_frag(&Ps[wid][lrow][kb]);
#pragma unroll
      for (int j = 0; j < 4; ++j) {
        v16bf vf = load_frag(&Vs[cur][j * 16 + lrow][kb]);
        o[j] = wmma_bf16(pf, vf, o[j]);
      }
    }
    __syncthreads();
  }

  const int b = bh / HEADS, h = bh % HEADS;
#pragma unroll
  for (int j = 0; j < 4; ++j) {
#pragma unroll
    for (int r = 0; r < 8; ++r) {
      int m   = q0 + r + rowadd;
      int col = h * DK + j * 16 + (lane & 15);
      float v = o[j][r] / lst[r];
      Ab[(size_t)(b * SEQ + m) * DMODEL + col] = (bf16)v;
    }
  }
}

// ---------------------------------------------------------------- launch
extern "C" void kernel_launch(void* const* d_in, const int* in_sizes, int n_in,
                              void* d_out, int out_size, void* d_ws, size_t ws_size,
                              hipStream_t stream) {
  (void)in_sizes; (void)n_in; (void)out_size; (void)ws_size;
  const float* x  = (const float*)d_in[0];
  const float* Wq = (const float*)d_in[1];
  const float* Wk = (const float*)d_in[2];
  const float* Wv = (const float*)d_in[3];
  const float* Wo = (const float*)d_in[4];
  const int*  pos = (const int*)d_in[5];
  float* out = (float*)d_out;

  char* ws = (char*)d_ws;
  const size_t MB = 1u << 20;
  bf16* xb  = (bf16*)(ws + 0);
  bf16* Wqb = (bf16*)(ws + 8 * MB);
  bf16* Wkb = (bf16*)(ws + 10 * MB);
  bf16* Wvb = (bf16*)(ws + 12 * MB);
  bf16* Wob = (bf16*)(ws + 14 * MB);
  bf16* Qb  = (bf16*)(ws + 16 * MB);
  bf16* Kb  = (bf16*)(ws + 24 * MB);
  bf16* Vt  = (bf16*)(ws + 32 * MB);
  bf16* Ab  = (bf16*)(ws + 40 * MB);

  cast_kernel<<<4096, 256, 0, stream>>>(x,  xb,  (MTOT * DMODEL) / 4);
  cast_kernel<<<1024, 256, 0, stream>>>(Wq, Wqb, (DMODEL * DMODEL) / 4);
  cast_kernel<<<1024, 256, 0, stream>>>(Wk, Wkb, (DMODEL * DMODEL) / 4);
  cast_kernel<<<1024, 256, 0, stream>>>(Wv, Wvb, (DMODEL * DMODEL) / 4);
  cast_kernel<<<1024, 256, 0, stream>>>(Wo, Wob, (DMODEL * DMODEL) / 4);

  dim3 gemm_grid(DMODEL / 128, MTOT / 128);
  gemm_wmma<0><<<gemm_grid, 256, 0, stream>>>(xb, Wqb, Qb, nullptr);
  gemm_wmma<0><<<gemm_grid, 256, 0, stream>>>(xb, Wkb, Kb, nullptr);
  gemm_wmma<1><<<gemm_grid, 256, 0, stream>>>(xb, Wvb, Vt, nullptr);

  rope_kernel<<<8192, 256, 0, stream>>>(Qb, pos);
  rope_kernel<<<8192, 256, 0, stream>>>(Kb, pos);

  attn_kernel<<<dim3(SEQ / 64, BATCH * HEADS), 128, 0, stream>>>(Qb, Kb, Vt, Ab);

  gemm_wmma<2><<<gemm_grid, 256, 0, stream>>>(Ab, Wob, nullptr, out);
}
